// MultiheadAttention_27986006901076
// MI455X (gfx1250) — matrix operand order repaired and checked by reading, not compiled, not run on hardware
//
#include <hip/hip_runtime.h>
#include <hip/hip_bf16.h>
#include <math.h>

// ---------------------------------------------------------------------------
// CDNA5 (gfx1250) fused transformer block, wave32 + v_wmma_f32_16x16x32_bf16
// Double-buffered LDS staging; attention K/V tiles streamed with
// GLOBAL_LOAD_ASYNC_TO_LDS (ASYNCcnt) when the toolchain exposes it.
// ---------------------------------------------------------------------------

typedef __bf16 bf16_t;
typedef __attribute__((ext_vector_type(4)))  __bf16 v4bf;
typedef __attribute__((ext_vector_type(8)))  __bf16 v8bf;
typedef __attribute__((ext_vector_type(16))) __bf16 v16bf;
typedef __attribute__((ext_vector_type(8)))  float  v8f;
typedef __attribute__((ext_vector_type(4)))  int    v4i;

#define WMMA_BF16(a, b, c) \
  __builtin_amdgcn_wmma_f32_16x16x32_bf16(false, (a), false, (b), (short)0, (c), false, false)

#define CAT16(lo, hi) \
  __builtin_shufflevector((lo), (hi), 0, 1, 2, 3, 4, 5, 6, 7, 8, 9, 10, 11, 12, 13, 14, 15)

#if __has_builtin(__builtin_amdgcn_global_load_async_to_lds_b128)
#define HAVE_ASYNC_LDS 1
#define ASYNC_CP_B128(g, l) \
  __builtin_amdgcn_global_load_async_to_lds_b128((v4i*)(g), (v4i*)(l), 0, 0)
#else
#define HAVE_ASYNC_LDS 0
#endif

#if __has_builtin(__builtin_amdgcn_s_wait_asynccnt)
#define WAIT_ASYNC() __builtin_amdgcn_s_wait_asynccnt(0)
#else
#define WAIT_ASYNC() asm volatile("s_wait_asynccnt 0x0" ::: "memory")
#endif

// A-matrix (16x32 bf16, ISA 7.12.2): lane-half h needs K = {8h..8h+7, 16+8h..16+8h+7}.
// Permute K within a row so each half's 16 values are contiguous.
__device__ __forceinline__ int a_slot(int k) {
  if (k < 8)  return k;
  if (k < 16) return k + 8;
  if (k < 24) return k - 8;
  return k;
}

// ---------------------------------------------------------------------------
// GEMM: out = act(A[MxK] @ W[KxN] + bias) (+ residual)
// block = 128 threads (4 waves), block tile 64x64, wave tile 16x64
// Double-buffered LDS: global loads of tile k+1 overlap WMMA on tile k.
// vtmode: write bf16 output transposed per attention head: [b][h][d][key]
// ---------------------------------------------------------------------------
__global__ __launch_bounds__(128)
void gemm_bias_kernel(const float* __restrict__ A, const float* __restrict__ W,
                      const float* __restrict__ bias, const float* __restrict__ resid,
                      float* __restrict__ outf, bf16_t* __restrict__ outb,
                      int M, int N, int K, int relu, int vtmode) {
  __shared__ bf16_t sA[2][64][40];  // A tile 64x32, K-slot-permuted, row pad->40
  __shared__ bf16_t sB[2][64][40];  // W tile stored TRANSPOSED: sB[p][col][k]

  const int row0 = blockIdx.y * 64;
  const int col0 = blockIdx.x * 64;
  const int t    = threadIdx.x;
  const int wave = t >> 5, lane = t & 31, half = lane >> 4, l16 = lane & 15;

  v8f acc[4] = {};
  float4 fa[4], fb[4];

  auto load_tile = [&](int k0) {  // batch all 8 b128 loads, no intermediate waits
#pragma unroll
    for (int j = 0; j < 4; ++j) {
      int c = t + 128 * j;
      fa[j] = *reinterpret_cast<const float4*>(
          A + (size_t)(row0 + (c >> 3)) * K + k0 + (c & 7) * 4);
      fb[j] = *reinterpret_cast<const float4*>(
          W + (size_t)(k0 + (c >> 4)) * N + col0 + (c & 15) * 4);
    }
  };
  auto store_tile = [&](int p) {
#pragma unroll
    for (int j = 0; j < 4; ++j) {
      int c = t + 128 * j;
      {
        v4bf pk = { (bf16_t)fa[j].x, (bf16_t)fa[j].y, (bf16_t)fa[j].z, (bf16_t)fa[j].w };
        *(v4bf*)&sA[p][c >> 3][a_slot((c & 7) * 4)] = pk;
      }
      {
        int rb = c >> 4, cb = (c & 15) * 4;
        sB[p][cb + 0][rb] = (bf16_t)fb[j].x;
        sB[p][cb + 1][rb] = (bf16_t)fb[j].y;
        sB[p][cb + 2][rb] = (bf16_t)fb[j].z;
        sB[p][cb + 3][rb] = (bf16_t)fb[j].w;
      }
    }
  };

  load_tile(0);
  store_tile(0);
  __syncthreads();

  for (int k0 = 0; k0 < K; k0 += 32) {
    const int p = (k0 >> 5) & 1;
    const bool more = (k0 + 32 < K);
    if (more) load_tile(k0 + 32);  // overlap with WMMA below

    v8bf alo = *(const v8bf*)&sA[p][wave * 16 + l16][half * 16];
    v8bf ahi = *(const v8bf*)&sA[p][wave * 16 + l16][half * 16 + 8];
    v16bf af = CAT16(alo, ahi);
#pragma unroll
    for (int ct = 0; ct < 4; ++ct) {
      v8bf blo = *(const v8bf*)&sB[p][ct * 16 + l16][half * 8];
      v8bf bhi = *(const v8bf*)&sB[p][ct * 16 + l16][half * 8 + 16];
      v16bf bfg = CAT16(blo, bhi);
      acc[ct] = WMMA_BF16(af, bfg, acc[ct]);
    }
    if (more) store_tile(1 - p);
    __syncthreads();
  }

  // C layout: VGPR r -> row r + 8*half, col = lane%16
#pragma unroll
  for (int ct = 0; ct < 4; ++ct) {
#pragma unroll
    for (int r = 0; r < 8; ++r) {
      int row = row0 + wave * 16 + r + 8 * half;
      int col = col0 + ct * 16 + l16;
      float vv = acc[ct][r] + bias[col];
      if (relu) vv = fmaxf(vv, 0.0f);
      if (resid) vv += resid[(size_t)row * N + col];
      if (outf) outf[(size_t)row * N + col] = vv;
      if (outb) {
        if (vtmode) {
          int b = row >> 11, key = row & 2047;  // row = b*2048 + key
          int h = col >> 6,  d   = col & 63;    // col = h*64 + d
          outb[(((size_t)(b * 16 + h) * 64 + d) << 11) + key] = (bf16_t)vv;
        } else {
          outb[(size_t)row * N + col] = (bf16_t)vv;
        }
      }
    }
  }
}

// ---------------------------------------------------------------------------
// Flash attention: 1 wave per (b, h, 16 q-rows); online softmax over 32-key
// tiles; S and PV via WMMA. K/V tiles streamed to LDS with async copies,
// double-buffered against the WMMA work. Head dim = 64. V pre-transposed.
// ---------------------------------------------------------------------------
__global__ __launch_bounds__(32)
void flash_attn_kernel(const bf16_t* __restrict__ Q, const bf16_t* __restrict__ Kk,
                       const bf16_t* __restrict__ Vt, const unsigned char* __restrict__ mask,
                       float* __restrict__ O) {
  constexpr int SQc = 2048, SKc = 2048, Dc = 1024, DH = 64, Hc = 16;
  const int qt = blockIdx.x, h = blockIdx.y, b = blockIdx.z;
  const int lane = threadIdx.x, half = lane >> 4, l16 = lane & 15;
  const float sm_scale = 0.03125f;  // 1/sqrt(1024)

  __shared__ bf16_t sK[2][32][64];  // 32 keys x 64 head-dims (4 KB / buf)
  __shared__ bf16_t sV[2][64][32];  // 64 head-dims x 32 keys (4 KB / buf)
  __shared__ bf16_t sP[16][40];     // P in A-slot-permuted layout

  // per-lane 16B copy assignments for tile staging
  const int krow = lane >> 2, kchk = (lane & 3) * 8;   // K: 8 rows x 4 chunks / op
  const int vrow = lane >> 2, vchk = (lane & 3) * 8;   // V: 8 rows x 4 chunks / op

  auto stage = [&](int kt, int p) {
#pragma unroll
    for (int i = 0; i < 4; ++i) {  // K tile: 32 rows of 128B
      const bf16_t* g = Kk + (size_t)(b * SKc + kt + krow + 8 * i) * Dc + h * DH + kchk;
#if HAVE_ASYNC_LDS
      ASYNC_CP_B128(g, &sK[p][krow + 8 * i][kchk]);
#else
      *(v8bf*)&sK[p][krow + 8 * i][kchk] = *(const v8bf*)g;
#endif
    }
#pragma unroll
    for (int i = 0; i < 8; ++i) {  // V tile: 64 rows of 64B
      const bf16_t* g = Vt + (((size_t)(b * Hc + h) * DH + vrow + 8 * i) << 11) + kt + vchk;
#if HAVE_ASYNC_LDS
      ASYNC_CP_B128(g, &sV[p][vrow + 8 * i][vchk]);
#else
      *(v8bf*)&sV[p][vrow + 8 * i][vchk] = *(const v8bf*)g;
#endif
    }
  };

  // Q fragments: two contiguous 16B runs per chunk (K = {8h..}, {16+8h..})
  v16bf qf[2];
  const bf16_t* qbase = Q + ((size_t)(b * SQc + qt * 16 + l16)) * Dc + h * DH;
#pragma unroll
  for (int c = 0; c < 2; ++c) {
    v8bf lo = *(const v8bf*)(qbase + c * 32 + 8 * half);
    v8bf hi = *(const v8bf*)(qbase + c * 32 + 16 + 8 * half);
    qf[c] = CAT16(lo, hi);
  }

  v8f o[4] = {};
  float mrow[8], lrow[8];
#pragma unroll
  for (int r = 0; r < 8; ++r) { mrow[r] = -INFINITY; lrow[r] = 0.f; }

  stage(0, 0);
  WAIT_ASYNC();

  for (int kt = 0; kt < SKc; kt += 32) {
    const int p = (kt >> 5) & 1;
    const bool more = (kt + 32 < SKc);
    if (more) stage(kt + 32, 1 - p);  // stream next tile while computing

    // --- S = (Q @ K^T) * scale for two 16-key subtiles (K frags from LDS) ---
    v8f s[2] = {};
#pragma unroll
    for (int st = 0; st < 2; ++st) {
      const bf16_t* kb = &sK[p][st * 16 + l16][0];
#pragma unroll
      for (int c = 0; c < 2; ++c) {
        v8bf lo = *(const v8bf*)(kb + c * 32 + 8 * half);
        v8bf hi = *(const v8bf*)(kb + c * 32 + 8 * half + 16);
        v16bf bfrag = CAT16(lo, hi);
        s[st] = WMMA_BF16(qf[c], bfrag, s[st]);
      }
      bool mv = mask[b * SKc + kt + st * 16 + l16] != 0;
#pragma unroll
      for (int r = 0; r < 8; ++r) {
        float x = s[st][r] * sm_scale;
        s[st][r] = mv ? x : -INFINITY;
      }
    }

    // --- online softmax row stats; write P straight into A-slot layout ---
    const int pslot = ((l16 >> 3) << 4) + (l16 & 7);
    float rescale[8];
#pragma unroll
    for (int r = 0; r < 8; ++r) {
      float v = fmaxf(s[0][r], s[1][r]);
#pragma unroll
      for (int off = 1; off < 16; off <<= 1) v = fmaxf(v, __shfl_xor(v, off, 32));
      float mn  = fmaxf(mrow[r], v);
      float scl = (mn == -INFINITY) ? 1.0f : __expf(mrow[r] - mn);
      float p0  = (mn == -INFINITY) ? 0.f : __expf(s[0][r] - mn);
      float p1  = (mn == -INFINITY) ? 0.f : __expf(s[1][r] - mn);
      float ps  = p0 + p1;
#pragma unroll
      for (int off = 1; off < 16; off <<= 1) ps += __shfl_xor(ps, off, 32);
      lrow[r] = lrow[r] * scl + ps;
      mrow[r] = mn;
      rescale[r] = scl;
      sP[r + 8 * half][pslot]     = (bf16_t)p0;
      sP[r + 8 * half][pslot + 8] = (bf16_t)p1;
    }

#pragma unroll
    for (int ct = 0; ct < 4; ++ct)
#pragma unroll
      for (int r = 0; r < 8; ++r) o[ct][r] *= rescale[r];

    asm volatile("s_wait_dscnt 0x0" ::: "memory");  // sP writes -> reads (1 wave)

    v8bf plo = *(const v8bf*)&sP[l16][half * 16];
    v8bf phi = *(const v8bf*)&sP[l16][half * 16 + 8];
    v16bf pf = CAT16(plo, phi);

    // --- O += P @ V (V frags contiguous in LDS) ---
#pragma unroll
    for (int ct = 0; ct < 4; ++ct) {
      const bf16_t* vb = &sV[p][ct * 16 + l16][8 * half];
      v8bf vlo = *(const v8bf*)vb;
      v8bf vhi = *(const v8bf*)(vb + 16);
      v16bf vfrag = CAT16(vlo, vhi);
      o[ct] = WMMA_BF16(pf, vfrag, o[ct]);
    }

    if (more) WAIT_ASYNC();  // next tile resident before we flip buffers
  }

  // normalize by row sums, 0 where fully masked (matches NaN->0 in reference)
#pragma unroll
  for (int ct = 0; ct < 4; ++ct)
#pragma unroll
    for (int r = 0; r < 8; ++r) {
      float denom = lrow[r];
      float vv = (denom > 0.f) ? o[ct][r] / denom : 0.f;
      int row = qt * 16 + r + 8 * half;
      O[((size_t)(b * SQc + row)) * Dc + h * DH + ct * 16 + l16] = vv;
    }
}

// ---------------------------------------------------------------------------
// out = LayerNorm(x (+ resid)) * g + b   (biased variance, eps = 1e-6)
// ---------------------------------------------------------------------------
__global__ __launch_bounds__(256)
void add_ln_kernel(const float* __restrict__ x, const float* __restrict__ rsd,
                   const float* __restrict__ g, const float* __restrict__ bta,
                   float* __restrict__ out, int Dn) {
  const int row = blockIdx.x;
  const int t = threadIdx.x;
  const float* xr = x + (size_t)row * Dn;
  const float* rr = rsd ? rsd + (size_t)row * Dn : nullptr;
  float v[4];
  float s = 0.f, ss = 0.f;
#pragma unroll
  for (int j = 0; j < 4; ++j) {
    int idx = t + 256 * j;
    float vv = xr[idx];
    if (rr) vv += rr[idx];
    v[j] = vv; s += vv; ss += vv * vv;
  }
#pragma unroll
  for (int off = 1; off < 32; off <<= 1) {
    s  += __shfl_xor(s, off, 32);
    ss += __shfl_xor(ss, off, 32);
  }
  __shared__ float shs[8], shss[8];
  int wv = t >> 5;
  if ((t & 31) == 0) { shs[wv] = s; shss[wv] = ss; }
  __syncthreads();
  s = 0.f; ss = 0.f;
#pragma unroll
  for (int w2 = 0; w2 < 8; ++w2) { s += shs[w2]; ss += shss[w2]; }
  float mean = s / Dn;
  float var  = ss / Dn - mean * mean;
  float rstd = rsqrtf(var + 1e-6f);
#pragma unroll
  for (int j = 0; j < 4; ++j) {
    int idx = t + 256 * j;
    out[(size_t)row * Dn + idx] = (v[j] - mean) * rstd * g[idx] + bta[idx];
  }
}

// ---------------------------------------------------------------------------
extern "C" void kernel_launch(void* const* d_in, const int* in_sizes, int n_in,
                              void* d_out, int out_size, void* d_ws, size_t ws_size,
                              hipStream_t stream) {
  (void)in_sizes; (void)n_in; (void)out_size; (void)ws_size;
  const float* q  = (const float*)d_in[0];
  const float* k  = (const float*)d_in[1];
  const float* v  = (const float*)d_in[2];
  const unsigned char* mask = (const unsigned char*)d_in[3];
  const float* Wq = (const float*)d_in[4];  const float* bq = (const float*)d_in[5];
  const float* Wk = (const float*)d_in[6];  const float* bk = (const float*)d_in[7];
  const float* Wv = (const float*)d_in[8];  const float* bv = (const float*)d_in[9];
  const float* Wo = (const float*)d_in[10]; const float* bo = (const float*)d_in[11];
  const float* g1 = (const float*)d_in[12]; const float* b1 = (const float*)d_in[13];
  const float* g2 = (const float*)d_in[14]; const float* b2 = (const float*)d_in[15];
  float* out = (float*)d_out;

  const int B = 4, SQ = 2048, SK = 2048, Dm = 1024, H = 16;
  const size_t MQ = (size_t)B * SQ;  // 8192 rows

  // workspace layout (~176 MB)
  char* ws = (char*)d_ws;
  float*  qp   = (float*)ws;  ws += MQ * Dm * sizeof(float);
  bf16_t* qpb  = (bf16_t*)ws; ws += MQ * Dm * sizeof(bf16_t);
  bf16_t* kpb  = (bf16_t*)ws; ws += MQ * Dm * sizeof(bf16_t);
  bf16_t* vtb  = (bf16_t*)ws; ws += MQ * Dm * sizeof(bf16_t);  // V^T [b][h][d][key]
  float*  attn = (float*)ws;  ws += MQ * Dm * sizeof(float);
  float*  xbuf = (float*)ws;  ws += MQ * Dm * sizeof(float);
  float*  zbuf = (float*)ws;  ws += MQ * Dm * sizeof(float);

  dim3 gg(Dm / 64, (unsigned)(MQ / 64));
  // QKV projections (qp kept fp32 for residual; bf16 copies feed attention)
  gemm_bias_kernel<<<gg, 128, 0, stream>>>(q, Wq, bq, nullptr, qp, qpb,
                                           (int)MQ, Dm, Dm, 0, 0);
  gemm_bias_kernel<<<gg, 128, 0, stream>>>(k, Wk, bk, nullptr, nullptr, kpb,
                                           (int)MQ, Dm, Dm, 0, 0);
  gemm_bias_kernel<<<gg, 128, 0, stream>>>(v, Wv, bv, nullptr, nullptr, vtb,
                                           (int)MQ, Dm, Dm, 0, 1);
  // masked multi-head attention (flash-style)
  dim3 ga(SQ / 16, H, B);
  flash_attn_kernel<<<ga, 32, 0, stream>>>(qpb, kpb, vtb, mask, attn);
  // x = LN(qp + attn) * g1 + b1
  add_ln_kernel<<<(unsigned)MQ, 256, 0, stream>>>(qp, attn, g1, b1, xbuf, Dm);
  // z = x + relu(x @ Wo + bo)
  gemm_bias_kernel<<<gg, 128, 0, stream>>>(xbuf, Wo, bo, xbuf, zbuf, nullptr,
                                           (int)MQ, Dm, Dm, 1, 0);
  // out = LN(z) * g2 + b2
  add_ln_kernel<<<(unsigned)MQ, 256, 0, stream>>>(zbuf, nullptr, g2, b2, out, Dm);
}
